// E62cTiedSelectiveCell_4887672783139
// MI455X (gfx1250) — matrix operand order repaired and checked by reading, not compiled, not run on hardware
//
#include <hip/hip_runtime.h>

// ---------------------------------------------------------------------------
// Problem dims (fixed by the reference):  x[T,B,D] @ W[D,D]^T + b  -> proj,
// then a sequential scan over T.
// ---------------------------------------------------------------------------
#define T_DIM 2048
#define B_DIM 8
#define D_DIM 1024
#define M_DIM (T_DIM * B_DIM)   // 16384 rows of the GEMM
#define K_DIM D_DIM              // 1024
#define N_DIM D_DIM              // 1024
#define BD    (B_DIM * D_DIM)    // 8192 independent scan chains

typedef __attribute__((ext_vector_type(16))) __bf16 v16bf;
typedef __attribute__((ext_vector_type(8)))  __bf16 v8bf;
typedef __attribute__((ext_vector_type(4)))  __bf16 v4bf;
typedef __attribute__((ext_vector_type(8)))  float  v8f;

union FragA {
    v16bf v;
    v8bf  h[2];
};

// ---------------------------------------------------------------------------
// Kernel 1: cast x (fp32 -> bf16) and W (fp32 -> bf16) into workspace,
// 4 elements per iteration (float4 in, 8 bytes of bf16 out).
// x_bf16 (32 MB) + W_bf16 (2 MB) fit in the 192 MB L2, so all GEMM fragment
// re-reads are L2 hits.
// ---------------------------------------------------------------------------
__global__ void cvt_bf16_kernel(const float4* __restrict__ x4,
                                const float4* __restrict__ W4,
                                v4bf* __restrict__ xb4,
                                v4bf* __restrict__ wb4) {
    const int NX4 = (M_DIM * K_DIM) / 4;    // 4,194,304
    const int NW4 = (N_DIM * K_DIM) / 4;    //   262,144
    const int total = NX4 + NW4;
    int stride = gridDim.x * blockDim.x;
    for (int i = blockIdx.x * blockDim.x + threadIdx.x; i < total; i += stride) {
        if (i < NX4) {
            float4 f = x4[i];
            v4bf o;
            o[0] = (__bf16)f.x; o[1] = (__bf16)f.y;
            o[2] = (__bf16)f.z; o[3] = (__bf16)f.w;
            xb4[i] = o;
        } else {
            float4 f = W4[i - NX4];
            v4bf o;
            o[0] = (__bf16)f.x; o[1] = (__bf16)f.y;
            o[2] = (__bf16)f.z; o[3] = (__bf16)f.w;
            wb4[i - NX4] = o;
        }
    }
}

// ---------------------------------------------------------------------------
// Kernel 2: proj[m, n] = sum_k xb[m, k] * wb[n, k] + bias[n]
// Both operands are row-major with K contiguous ("NT" GEMM) which matches the
// WMMA A (16x32, M-major) and B (32x16, N-major) bf16 fragment layouts.
//
// Block: 256 threads = 8 waves (4 along M x 2 along N) -> 128(M) x 128(N).
// Wave: 32(M) x 64(N) output = 2x4 WMMA fragments, K-loop step 32:
//       12 x b128 loads feeding 8 x v_wmma_f32_16x16x32_bf16 (1.5:1).
//
// Fragment layouts used here (A/B in this code's contiguous k order, same for both; C/D as the ISA gives):
//   A 16x32 bf16: lane l (row = l&15, half = l>>4) holds
//                 K = {half*16 .. half*16+7} in v0..v3 and
//                 K = {half*16+8 ..}         in v4..v7      -> two 16B loads
//   B 32x16 bf16: lane l (col = l&15, half = l>>4) holds
//                 K = half*16 .. half*16+15 contiguous      -> one 32B load
//   C/D f32 16x16: VGPR v, lane l -> (M = v + (l>>4)*8, N = l&15)
// ---------------------------------------------------------------------------
__global__ __launch_bounds__(256) void gemm_wmma_kernel(
        const __bf16* __restrict__ xb,
        const __bf16* __restrict__ wb,
        const float*  __restrict__ bias,
        float*        __restrict__ proj) {
    const int wave = threadIdx.x >> 5;
    const int lane = threadIdx.x & 31;
    const int r    = lane & 15;
    const int half = lane >> 4;

    const int wm = wave & 3;          // 4 waves along M (32 rows each)
    const int wn = wave >> 2;         // 2 waves along N (64 cols each)
    const int Mbase = blockIdx.x * 128 + wm * 32;
    const int Nbase = blockIdx.y * 128 + wn * 64;

    v8f c00 = {}, c01 = {}, c02 = {}, c03 = {};
    v8f c10 = {}, c11 = {}, c12 = {}, c13 = {};

    const __bf16* arow0 = xb + (size_t)(Mbase + r) * K_DIM;
    const __bf16* arow1 = arow0 + (size_t)16 * K_DIM;
    const __bf16* brow0 = wb + (size_t)(Nbase + r) * K_DIM;
    const __bf16* brow1 = brow0 + (size_t)16 * K_DIM;
    const __bf16* brow2 = brow0 + (size_t)32 * K_DIM;
    const __bf16* brow3 = brow0 + (size_t)48 * K_DIM;

    for (int k = 0; k < K_DIM; k += 32) {
        FragA a0, a1;
        a0.h[0] = *(const v8bf*)(arow0 + k + half * 16);
        a0.h[1] = *(const v8bf*)(arow0 + k + 8 + half * 16);
        a1.h[0] = *(const v8bf*)(arow1 + k + half * 16);
        a1.h[1] = *(const v8bf*)(arow1 + k + 8 + half * 16);
        v16bf b0 = *(const v16bf*)(brow0 + k + half * 16);
        v16bf b1 = *(const v16bf*)(brow1 + k + half * 16);
        v16bf b2 = *(const v16bf*)(brow2 + k + half * 16);
        v16bf b3 = *(const v16bf*)(brow3 + k + half * 16);

        // 8 args: (neg_a, A, neg_b, B, c_mod, C, reuse_a, reuse_b)
        c00 = __builtin_amdgcn_wmma_f32_16x16x32_bf16(false, a0.v, false, b0,
                                                      (short)0, c00, false, false);
        c01 = __builtin_amdgcn_wmma_f32_16x16x32_bf16(false, a0.v, false, b1,
                                                      (short)0, c01, false, false);
        c02 = __builtin_amdgcn_wmma_f32_16x16x32_bf16(false, a0.v, false, b2,
                                                      (short)0, c02, false, false);
        c03 = __builtin_amdgcn_wmma_f32_16x16x32_bf16(false, a0.v, false, b3,
                                                      (short)0, c03, false, false);
        c10 = __builtin_amdgcn_wmma_f32_16x16x32_bf16(false, a1.v, false, b0,
                                                      (short)0, c10, false, false);
        c11 = __builtin_amdgcn_wmma_f32_16x16x32_bf16(false, a1.v, false, b1,
                                                      (short)0, c11, false, false);
        c12 = __builtin_amdgcn_wmma_f32_16x16x32_bf16(false, a1.v, false, b2,
                                                      (short)0, c12, false, false);
        c13 = __builtin_amdgcn_wmma_f32_16x16x32_bf16(false, a1.v, false, b3,
                                                      (short)0, c13, false, false);
    }

    const float bias0 = bias[Nbase + r];
    const float bias1 = bias[Nbase + 16 + r];
    const float bias2 = bias[Nbase + 32 + r];
    const float bias3 = bias[Nbase + 48 + r];

#pragma unroll
    for (int v = 0; v < 8; ++v) {
        const int m0 = Mbase + half * 8 + v;
        const int m1 = m0 + 16;
        float* row0 = proj + (size_t)m0 * N_DIM + Nbase + r;
        float* row1 = proj + (size_t)m1 * N_DIM + Nbase + r;
        row0[0]  = c00[v] + bias0;
        row0[16] = c01[v] + bias1;
        row0[32] = c02[v] + bias2;
        row0[48] = c03[v] + bias3;
        row1[0]  = c10[v] + bias0;
        row1[16] = c11[v] + bias1;
        row1[32] = c12[v] + bias2;
        row1[48] = c13[v] + bias3;
    }
}

// ---------------------------------------------------------------------------
// Kernel 3: sequential scan. One thread per (b, d) chain (8192 chains).
//   alpha = sigmoid(p); h = alpha*h + (1-alpha)*tanh(p); out = h*silu(h)
//
// The only true serial dependency is ONE fma per step:
//   h = fma(alpha_t, h, beta_t)  with  beta_t = (1-alpha_t)*tanh(p_t),
// since alpha/beta depend only on p and the output h*silu(h) doesn't feed
// back. So: unroll by 4, batch the 4 loads, compute the 4 (alpha, beta)
// gate pairs as independent ILP, run the 4-fma chain, then compute/store the
// 4 outputs independently. Divides are replaced with v_rcp_f32.
//
// Gates fused through one exp: e = e^p; sigmoid = e/(e+1);
// tanh = (e^2-1)/(e^2+1). Reads proj[t] from `outbuf` and overwrites it in
// place with out[t] (same thread, read-before-write). h[0] = h0.
// ---------------------------------------------------------------------------
__global__ __launch_bounds__(256) void scan_kernel(const float* __restrict__ h0,
                                                   float* __restrict__ outbuf,
                                                   float* __restrict__ hbuf) {
    const int i = blockIdx.x * blockDim.x + threadIdx.x;
    if (i >= BD) return;

    float h = h0[i];
    hbuf[i] = h;                           // h[0]

    float* p  = outbuf + i;
    float* hp = hbuf + BD + i;             // h[1..T]

    for (int t = 0; t < T_DIM; t += 4) {
        // ---- 4 independent loads ----
        float pv[4];
#pragma unroll
        for (int u = 0; u < 4; ++u)
            pv[u] = p[(size_t)(t + u) * BD];

        // ---- 4 independent gate computations ----
        float alpha[4], beta[4];
#pragma unroll
        for (int u = 0; u < 4; ++u) {
            float e  = __expf(pv[u]);
            float a  = e * __builtin_amdgcn_rcpf(e + 1.0f);          // sigmoid(p)
            float e2 = e * e;
            float th = (e2 - 1.0f) * __builtin_amdgcn_rcpf(e2 + 1.0f); // tanh(p)
            alpha[u] = a;
            beta[u]  = (1.0f - a) * th;
        }

        // ---- serial recurrence: 4 dependent fmas ----
        float hs[4];
#pragma unroll
        for (int u = 0; u < 4; ++u) {
            h = fmaf(alpha[u], h, beta[u]);
            hs[u] = h;
        }

        // ---- 4 independent outputs + stores ----
#pragma unroll
        for (int u = 0; u < 4; ++u) {
            float hv = hs[u];
            float eh = __expf(hv);
            float out = hv * hv * eh * __builtin_amdgcn_rcpf(eh + 1.0f); // h*silu(h)
            p[(size_t)(t + u) * BD]  = out;
            hp[(size_t)(t + u) * BD] = hv;
        }
    }
}

// ---------------------------------------------------------------------------
// Launcher. Inputs: x[T,B,D] f32, W[D,D] f32, b[D] f32, h0[B,D] f32.
// Output: outs [T,B,D] followed by h [T+1,B,D], both f32.
// Workspace: bf16 copies of x (32 MB) and W (2 MB).
// ---------------------------------------------------------------------------
extern "C" void kernel_launch(void* const* d_in, const int* in_sizes, int n_in,
                              void* d_out, int out_size, void* d_ws, size_t ws_size,
                              hipStream_t stream) {
    const float* x  = (const float*)d_in[0];
    const float* W  = (const float*)d_in[1];
    const float* b  = (const float*)d_in[2];
    const float* h0 = (const float*)d_in[3];

    float* outp = (float*)d_out;                       // outs region (also proj scratch)
    float* hout = outp + (size_t)T_DIM * BD;           // h region: (T+1)*B*D

    __bf16* xb = (__bf16*)d_ws;
    __bf16* wb = xb + (size_t)M_DIM * K_DIM;

    // 1) fp32 -> bf16 casts (vectorized x4)
    cvt_bf16_kernel<<<2048, 256, 0, stream>>>((const float4*)x, (const float4*)W,
                                              (v4bf*)xb, (v4bf*)wb);

    // 2) WMMA GEMM: proj = x @ W^T + b, written into the outs region
    dim3 grid(M_DIM / 128, N_DIM / 128);
    gemm_wmma_kernel<<<grid, 256, 0, stream>>>(xb, wb, b, outp);

    // 3) sequential scan + elementwise epilogue (in-place over proj)
    scan_kernel<<<BD / 256, 256, 0, stream>>>(h0, outp, hout);
}
